// ExactReactionDiffusion2D_37134287241540
// MI455X (gfx1250) — matrix operand order repaired and checked by reading, not compiled, not run on hardware
//
#include <hip/hip_runtime.h>
#include <hip/hip_bf16.h>
#include <math.h>

typedef __attribute__((ext_vector_type(16))) __bf16 v16bf;
typedef __attribute__((ext_vector_type(8)))  float  v8f;

#define B_SZ   8
#define N_TOK  4096
#define D_DIM  512
#define HW     64
#define KTOT   1024      // 2 * D_DIM
#define M_TOT  32768     // B * N
#define PITCH  65        // 64 + 1 pad: conflict-free column FFT passes

__device__ __forceinline__ float softplus_f(float z) {
  return (z > 20.f) ? z : log1pf(expf(z));
}

// ---------------- phase 1: pooled mean over tokens ----------------
__global__ void pool_kernel(const float* __restrict__ x, float* __restrict__ pooled) {
  int b = blockIdx.x, d = threadIdx.x;                 // 8 blocks x 512 threads
  const float* p = x + (size_t)b * N_TOK * D_DIM + d;  // coalesced across d
  float acc = 0.f;
  for (int n = 0; n < N_TOK; ++n) acc += p[(size_t)n * D_DIM];
  pooled[b * D_DIM + d] = acc * (1.0f / N_TOK);
}

// ---------------- phase 2: diffusion gate gamma ----------------
__global__ void gamma_kernel(const float* __restrict__ pooled, const float* __restrict__ Wg,
                             const float* __restrict__ bg, const float* __restrict__ log_base,
                             float* __restrict__ gam) {
  __shared__ float pl[D_DIM];
  int b = blockIdx.x, j = threadIdx.x;
  pl[j] = pooled[b * D_DIM + j];
  __syncthreads();
  const float* wr = Wg + (size_t)j * D_DIM;
  float s = 0.f;
  for (int k = 0; k < D_DIM; ++k) s = fmaf(pl[k], wr[k], s);
  s += bg[j] + log_base[j];
  gam[b * D_DIM + j] = fminf(softplus_f(s), 0.35f);
}

// ---------------- phase 3: x [B,N,D] -> xt [B,D,N] tiled transpose ----------------
__global__ void transpose_kernel(const float* __restrict__ x, float* __restrict__ xt) {
  __shared__ float t[32][33];
  int b = blockIdx.z;
  int n0 = blockIdx.x * 32, d0 = blockIdx.y * 32;
  for (int i = threadIdx.y; i < 32; i += 8)
    t[i][threadIdx.x] = x[((size_t)b * N_TOK + (n0 + i)) * D_DIM + d0 + threadIdx.x];
  __syncthreads();
  for (int i = threadIdx.y; i < 32; i += 8)
    xt[((size_t)b * D_DIM + (d0 + i)) * N_TOK + n0 + threadIdx.x] = t[threadIdx.x][i];
}

// ---------------- phase 4: Wout [o][k] fp32 -> bf16 hi/lo, same layout ----------------
__global__ void prep_wout_kernel(const float* __restrict__ Wout,
                                 __bf16* __restrict__ Whi, __bf16* __restrict__ Wlo) {
  int i = blockIdx.x * 256 + threadIdx.x;   // fully coalesced both sides
  float v = Wout[i];
  __bf16 h = (__bf16)v;
  Whi[i] = h;
  Wlo[i] = (__bf16)(v - (float)h);
}

// ---------------- in-LDS radix-2 FFT helpers (length 64, rows / cols) ----------------
__device__ __forceinline__ void fft_rows(float* re, float* im, const float* twr,
                                         const float* twi, float wsign, int tid) {
  __syncthreads();
  for (int i = tid; i < 4096; i += 256) {          // bit-reverse within each row
    int r = i >> 6, e = i & 63;
    int rv = __brev((unsigned)e) >> 26;
    if (e < rv) {
      int a = r * PITCH + e, b = r * PITCH + rv;
      float tr = re[a]; re[a] = re[b]; re[b] = tr;
      float ti = im[a]; im[a] = im[b]; im[b] = ti;
    }
  }
  for (int s = 1; s <= 6; ++s) {
    __syncthreads();
    int half = 1 << (s - 1);
    for (int i = tid; i < 2048; i += 256) {
      int r = i >> 5, j = i & 31;
      int grp = j >> (s - 1);
      int t = j & (half - 1);
      int pos = (grp << s) + t;
      int twix = t << (6 - s);
      float wr = twr[twix], wi = wsign * twi[twix];
      int ia = r * PITCH + pos, ib = ia + half;
      float ur = re[ia], ui = im[ia];
      float vr0 = re[ib], vi0 = im[ib];
      float vr = vr0 * wr - vi0 * wi;
      float vi = vr0 * wi + vi0 * wr;
      re[ia] = ur + vr; im[ia] = ui + vi;
      re[ib] = ur - vr; im[ib] = ui - vi;
    }
  }
}

__device__ __forceinline__ void fft_cols(float* re, float* im, const float* twr,
                                         const float* twi, float wsign, int tid) {
  __syncthreads();
  for (int i = tid; i < 4096; i += 256) {          // bit-reverse within each column
    int e = i >> 6, c = i & 63;
    int rv = __brev((unsigned)e) >> 26;
    if (e < rv) {
      int a = e * PITCH + c, b = rv * PITCH + c;
      float tr = re[a]; re[a] = re[b]; re[b] = tr;
      float ti = im[a]; im[a] = im[b]; im[b] = ti;
    }
  }
  for (int s = 1; s <= 6; ++s) {
    __syncthreads();
    int half = 1 << (s - 1);
    for (int i = tid; i < 2048; i += 256) {
      int c = i & 63, j = i >> 6;
      int grp = j >> (s - 1);
      int t = j & (half - 1);
      int pos = (grp << s) + t;
      int twix = t << (6 - s);
      float wr = twr[twix], wi = wsign * twi[twix];
      int ia = pos * PITCH + c, ib = ia + half * PITCH;
      float ur = re[ia], ui = im[ia];
      float vr0 = re[ib], vi0 = im[ib];
      float vr = vr0 * wr - vi0 * wi;
      float vi = vr0 * wi + vi0 * wr;
      re[ia] = ur + vr; im[ia] = ui + vi;
      re[ib] = ur - vr; im[ib] = ui - vi;
    }
  }
}

__device__ __forceinline__ void reaction(float* re, float* im, float a2, float b2,
                                         float expm, bool use_zero, float dtr,
                                         float pre, int tid) {
  __syncthreads();
  for (int i = tid; i < 4096; i += 256) {
    int p = (i >> 6) * PITCH + (i & 63);
    float pr = re[p] * pre, pi = im[p] * pre;
    float q = pr * pr + pi * pi;
    float denom = fmaxf(b2 * q + (a2 - b2 * q) * expm, 1e-8f);
    float qg = a2 * q / denom;
    float qz = q / (1.f + b2 * q * dtr);
    float qn = fmaxf(use_zero ? qz : qg, 0.f);
    float sc = sqrtf(qn / fmaxf(q, 1e-8f));
    re[p] = pr * sc;
    im[p] = pi * sc;
  }
}

// ---------------- phase 5: 4-step spectral reaction-diffusion per (b,d) field ----------------
__global__ void __launch_bounds__(256)
evolve_kernel(const float* __restrict__ xt, const float* __restrict__ gam,
              const float* __restrict__ alpha_raw, const float* __restrict__ beta_raw,
              __bf16* __restrict__ AhiT, __bf16* __restrict__ AloT) {
  __shared__ float re[HW * PITCH];
  __shared__ float im[HW * PITCH];
  __shared__ float twr[32], twi[32];
  __shared__ float pg[64];
  int tid = threadIdx.x;
  int bd = blockIdx.x;
  int b = bd >> 9, d = bd & 511;

  float g = gam[b * D_DIM + d];
  float alpha = 0.25f * tanhf(alpha_raw[d]);
  float beta = softplus_f(beta_raw[d]) + 1e-4f;
  float a2 = 2.f * alpha, b2 = 2.f * beta;
  const float dt = 0.25f, dtr = 0.125f;
  float expm = expf(-a2 * dtr);
  bool use_zero = fabsf(a2) < 1e-6f;

  if (tid < 32) {
    float ang = -6.283185307179586f * (float)tid / 64.f;   // forward twiddles e^{-i*2*pi*k/64}
    twr[tid] = cosf(ang);
    twi[tid] = sinf(ang);
  }
  if (tid < 64) {
    pg[tid] = dt * g * (2.f * cosf(6.283185307179586f * (float)tid / 64.f) - 2.f);
  }

  const float* src = xt + (size_t)bd * N_TOK;
  for (int i = tid; i < 4096; i += 256) {
    int p = (i >> 6) * PITCH + (i & 63);
    re[p] = src[i];
    im[p] = 0.f;
  }

  for (int step = 0; step < 4; ++step) {
    reaction(re, im, a2, b2, expm, use_zero, dtr, 1.f, tid);
    fft_rows(re, im, twr, twi, 1.f, tid);
    fft_cols(re, im, twr, twi, 1.f, tid);
    __syncthreads();
    for (int i = tid; i < 4096; i += 256) {     // spectral phase rotation
      int y = i >> 6, xx = i & 63;
      float arg = pg[y] + pg[xx];
      float s, c;
      __sincosf(arg, &s, &c);
      int p = y * PITCH + xx;
      float r0 = re[p], i0 = im[p];
      re[p] = r0 * c - i0 * s;
      im[p] = r0 * s + i0 * c;
    }
    fft_rows(re, im, twr, twi, -1.f, tid);
    fft_cols(re, im, twr, twi, -1.f, tid);
    reaction(re, im, a2, b2, expm, use_zero, dtr, 1.f / 4096.f, tid);  // fold ifft 1/N
  }

  __syncthreads();
  // emit combined matrix K-major as bf16 hi/lo: AT[k][m], k = d (real) / 512+d (imag)
  size_t mb = (size_t)b * N_TOK;
  for (int i = tid; i < 4096; i += 256) {
    int p = (i >> 6) * PITCH + (i & 63);
    float vr = re[p], vi = im[p];
    __bf16 hr = (__bf16)vr; __bf16 lr = (__bf16)(vr - (float)hr);
    __bf16 hi2 = (__bf16)vi; __bf16 li = (__bf16)(vi - (float)hi2);
    size_t m = mb + i;
    AhiT[(size_t)d * M_TOT + m] = hr;
    AloT[(size_t)d * M_TOT + m] = lr;
    AhiT[(size_t)(D_DIM + d) * M_TOT + m] = hi2;
    AloT[(size_t)(D_DIM + d) * M_TOT + m] = li;
  }
}

// ---------------- phase 5.5: bf16 tiled transpose AT[k][m] -> A2[m][k] (hi+lo) ----------------
__global__ void __launch_bounds__(256)
transpose_bf16_kernel(const __bf16* __restrict__ AhiT, const __bf16* __restrict__ AloT,
                      __bf16* __restrict__ A2hi, __bf16* __restrict__ A2lo) {
  __shared__ unsigned short th[64][72];   // 64(k) x 64(m) tile, pad 8
  __shared__ unsigned short tl[64][72];
  int tid = threadIdx.x;
  int k0 = blockIdx.x * 64, m0 = blockIdx.y * 64;
  const unsigned int* gh = (const unsigned int*)AhiT;
  const unsigned int* gl = (const unsigned int*)AloT;
  for (int i = tid; i < 2048; i += 256) {     // coalesced: m contiguous per k row
    int kk = i >> 5, mp = i & 31;
    size_t go = (((size_t)(k0 + kk) * M_TOT) >> 1) + (m0 >> 1) + mp;
    *(unsigned int*)&th[kk][2 * mp] = gh[go];
    *(unsigned int*)&tl[kk][2 * mp] = gl[go];
  }
  __syncthreads();
  unsigned int* oh = (unsigned int*)A2hi;
  unsigned int* ol = (unsigned int*)A2lo;
  for (int i = tid; i < 2048; i += 256) {     // coalesced: k contiguous per m row
    int mm = i >> 5, kp = i & 31;
    unsigned int h0 = th[2 * kp][mm], h1 = th[2 * kp + 1][mm];
    unsigned int l0 = tl[2 * kp][mm], l1 = tl[2 * kp + 1][mm];
    size_t go = (((size_t)(m0 + mm) * KTOT) >> 1) + (k0 >> 1) + kp;
    oh[go] = h0 | (h1 << 16);
    ol[go] = l0 | (l1 << 16);
  }
}

// ---------------- phase 6: split-bf16 WMMA GEMM, all-b128 LDS path ----------------
// out[m][o] = sum_k combined[m][k] * Wout[o][k] + x[m][o] * Dparam[o]
// A2 global [m][k] (k contiguous), W global [o][k] (k contiguous).
// Block tile 128(M) x 64(N), 8 waves, each wave owns 16x64 (4 accumulators).
#define LDA 40   // bf16 per LDS row (80B = 20 dwords; gcd(20,64)=4 -> spread banks)

__global__ void __launch_bounds__(256)
gemm_kernel(const __bf16* __restrict__ A2hi, const __bf16* __restrict__ A2lo,
            const __bf16* __restrict__ Whi, const __bf16* __restrict__ Wlo,
            const float* __restrict__ x, const float* __restrict__ Dparam,
            float* __restrict__ out) {
  __shared__ __align__(16) __bf16 Ah[128][LDA];
  __shared__ __align__(16) __bf16 Al[128][LDA];
  __shared__ __align__(16) __bf16 Bh[64][LDA];
  __shared__ __align__(16) __bf16 Bl[64][LDA];

  int tid = threadIdx.x;
  int lane = tid & 31, wave = tid >> 5;
  int m_base = blockIdx.x * 128;
  int n_base = blockIdx.y * 64;

  v8f acc[4];
#pragma unroll
  for (int t = 0; t < 4; ++t)
#pragma unroll
    for (int r = 0; r < 8; ++r) acc[t][r] = 0.f;

  int khi = (lane >= 16) ? 8 : 0;    // A: e<8 -> K=khi+e ; e>=8 -> K=khi+16+(e-8)
  int kbo = (lane >= 16) ? 8 : 0;    // B: e<8 -> K=kbo+e ; e>=8 -> K=kbo+16+(e-8)
  int mloc = wave * 16 + (lane & 15);

  for (int k0 = 0; k0 < KTOT; k0 += 32) {
    __syncthreads();
    // A tiles: 128 rows x 32 k = 512 b128 segments per array (2 per thread)
#pragma unroll
    for (int i = tid; i < 512; i += 256) {
      int m = i >> 2, seg = i & 3;
      size_t go = (size_t)(m_base + m) * KTOT + k0 + seg * 8;
      *(uint4*)&Ah[m][seg * 8] = *(const uint4*)(A2hi + go);
      *(uint4*)&Al[m][seg * 8] = *(const uint4*)(A2lo + go);
    }
    // B tiles: 64 rows x 32 k = 256 b128 segments per array (1 per thread)
    {
      int n = tid >> 2, seg = tid & 3;
      size_t go = (size_t)(n_base + n) * KTOT + k0 + seg * 8;
      *(uint4*)&Bh[n][seg * 8] = *(const uint4*)(Whi + go);
      *(uint4*)&Bl[n][seg * 8] = *(const uint4*)(Wlo + go);
    }
    __syncthreads();

    union { v16bf v; uint4 q[2]; } ah, al, bh, bl;
    ah.q[0] = *(const uint4*)&Ah[mloc][khi];        // K khi..khi+7
    ah.q[1] = *(const uint4*)&Ah[mloc][khi + 16];   // K khi+16..khi+23
    al.q[0] = *(const uint4*)&Al[mloc][khi];
    al.q[1] = *(const uint4*)&Al[mloc][khi + 16];
#pragma unroll
    for (int t = 0; t < 4; ++t) {
      int nloc = t * 16 + (lane & 15);
      bh.q[0] = *(const uint4*)&Bh[nloc][kbo];      // K kbo..kbo+7
      bh.q[1] = *(const uint4*)&Bh[nloc][kbo + 16]; // K kbo+16..kbo+23
      bl.q[0] = *(const uint4*)&Bl[nloc][kbo];
      bl.q[1] = *(const uint4*)&Bl[nloc][kbo + 16];
      // hi*hi + hi*lo + lo*hi -> near-fp32 product, f32 accumulate
      acc[t] = __builtin_amdgcn_wmma_f32_16x16x32_bf16(false, ah.v, false, bh.v,
                                                       (short)0, acc[t], false, false);
      acc[t] = __builtin_amdgcn_wmma_f32_16x16x32_bf16(false, ah.v, false, bl.v,
                                                       (short)0, acc[t], false, false);
      acc[t] = __builtin_amdgcn_wmma_f32_16x16x32_bf16(false, al.v, false, bh.v,
                                                       (short)0, acc[t], false, false);
    }
  }

  int mrow_off = (lane >= 16) ? 8 : 0;          // D layout: lane=N, VGPR r -> M=r(+8)
#pragma unroll
  for (int t = 0; t < 4; ++t) {
    int og = n_base + t * 16 + (lane & 15);
    float dp = Dparam[og];
#pragma unroll
    for (int r = 0; r < 8; ++r) {
      int mg = m_base + wave * 16 + r + mrow_off;
      size_t o = (size_t)mg * D_DIM + og;
      out[o] = acc[t][r] + x[o] * dp;
    }
  }
}

// ---------------- launcher ----------------
extern "C" void kernel_launch(void* const* d_in, const int* in_sizes, int n_in,
                              void* d_out, int out_size, void* d_ws, size_t ws_size,
                              hipStream_t stream) {
  (void)in_sizes; (void)n_in; (void)out_size; (void)ws_size;
  const float* x         = (const float*)d_in[0];
  const float* Wg        = (const float*)d_in[1];
  const float* bg        = (const float*)d_in[2];
  const float* log_base  = (const float*)d_in[3];
  const float* alpha_raw = (const float*)d_in[4];
  const float* beta_raw  = (const float*)d_in[5];
  const float* Wout      = (const float*)d_in[6];
  const float* Dparam    = (const float*)d_in[7];
  float* out = (float*)d_out;

  char* ws = (char*)d_ws;
  const size_t SZ_XT  = (size_t)B_SZ * D_DIM * N_TOK * sizeof(float);       // 64 MB
  const size_t SZ_AB  = (size_t)KTOT * M_TOT * sizeof(__bf16);              // 64 MB
  const size_t SZ_W   = (size_t)D_DIM * KTOT * sizeof(__bf16);              // 1 MB
  float*  xt     = (float*)ws;                     // consumed by evolve, then aliased
  __bf16* A2hi   = (__bf16*)ws;                    // alias over xt (stream-ordered WAR safe)
  float*  pooled = (float*)(ws + SZ_XT);
  float*  gam    = pooled + B_SZ * D_DIM;
  __bf16* Whi    = (__bf16*)(ws + SZ_XT + 65536);
  __bf16* Wlo    = Whi + (size_t)D_DIM * KTOT;
  __bf16* AhiT   = (__bf16*)(ws + SZ_XT + 65536 + 2 * SZ_W);
  __bf16* AloT   = AhiT + (size_t)KTOT * M_TOT;
  __bf16* A2lo   = AloT + (size_t)KTOT * M_TOT;
  (void)SZ_AB;

  pool_kernel<<<B_SZ, D_DIM, 0, stream>>>(x, pooled);
  gamma_kernel<<<B_SZ, D_DIM, 0, stream>>>(pooled, Wg, bg, log_base, gam);
  transpose_kernel<<<dim3(N_TOK / 32, D_DIM / 32, B_SZ), dim3(32, 8), 0, stream>>>(x, xt);
  prep_wout_kernel<<<(D_DIM * KTOT) / 256, 256, 0, stream>>>(Wout, Whi, Wlo);
  evolve_kernel<<<B_SZ * D_DIM, 256, 0, stream>>>(xt, gam, alpha_raw, beta_raw, AhiT, AloT);
  transpose_bf16_kernel<<<dim3(KTOT / 64, M_TOT / 64), 256, 0, stream>>>(AhiT, AloT, A2hi, A2lo);
  gemm_kernel<<<dim3(M_TOT / 128, D_DIM / 64), 256, 0, stream>>>(A2hi, A2lo, Whi, Wlo,
                                                                 x, Dparam, out);
}
